// ECG_XNOR0_48309792145685
// MI455X (gfx1250) — 8-bit matrix operand order repaired and checked by reading, compiled, not run on hardware
//
#include <hip/hip_runtime.h>
#include <cstdint>
#include <cstddef>

typedef int v8i __attribute__((ext_vector_type(8)));
typedef int v4i __attribute__((ext_vector_type(4)));
typedef int v2i __attribute__((ext_vector_type(2)));

#define EPS      1e-5f
#define BATCH    2048
#define KW       11
#define CPAD     5
#define MAXCI    72
#define KKPADMAX 832
#define MAXWAVES 5
#define PTILE    6

__device__ __forceinline__ int8_t sgn8(float v) {
  return v > 0.f ? (int8_t)1 : (v < 0.f ? (int8_t)-1 : (int8_t)0);
}

__global__ void binarize_weights_kernel(const float* __restrict__ w,
                                        int8_t* __restrict__ wb,
                                        int co, int co16, int KK, int KKpad) {
  int idx = blockIdx.x * blockDim.x + threadIdx.x;
  if (idx >= co16 * KKpad) return;
  int o = idx / KKpad, k = idx - o * KKpad;
  wb[idx] = (o < co && k < KK) ? sgn8(w[o * KK + k]) : (int8_t)0;
}

__global__ __launch_bounds__(MAXWAVES * 32)
void xnor_block_fused_kernel(const float* __restrict__ x,
                             const int8_t* __restrict__ wb,
                             const float* __restrict__ bias,
                             const float* __restrict__ g,
                             const float* __restrict__ beta,
                             const float* __restrict__ mu,
                             const float* __restrict__ var,
                             float* __restrict__ pout,
                             int ci, int co, int L, int Lp,
                             int KK, int KKpad) {
  __shared__ float  s_scale[MAXCI];
  __shared__ float  s_shift[MAXCI];
  __shared__ int8_t s_x[MAXCI * 32];
  __shared__ __align__(16) int8_t s_im[16 * KKPADMAX];
  __shared__ float  s_y[MAXWAVES][16][16];

  const int s    = blockIdx.y;
  const int P0   = blockIdx.x * PTILE;
  const int m0   = 2 * P0;
  const int tid  = (int)threadIdx.x;
  const int nthr = (int)blockDim.x;

  for (int c = tid; c < ci; c += nthr) {
    float sc = g[c] * rsqrtf(var[c] + EPS);
    s_scale[c] = sc;
    s_shift[c] = beta[c] - mu[c] * sc;
  }
  __syncthreads();

  for (int idx = tid; idx < ci * 32; idx += nthr) {
    int c = idx >> 5, p = idx & 31;
    int gp = m0 - CPAD + p;
    int8_t v = 0;
    if (p < 16 + 2 * CPAD && gp >= 0 && gp < L) {
      float f = x[((size_t)s * ci + c) * L + gp] * s_scale[c] + s_shift[c];
      v = sgn8(f);
    }
    s_x[idx] = v;
  }
  __syncthreads();

  for (int idx = tid; idx < (KKpad << 4); idx += nthr) {
    int k = idx >> 4, row = idx & 15;
    int8_t v = 0;
    if (k < KK) {
      int c = k / KW, t = k - c * KW;
      v = s_x[(c << 5) + row + t];
    }
    s_im[row * KKpad + k] = v;
  }
  __syncthreads();

  const int lane = tid & 31;
  const int col  = lane & 15;
  const int hi   = lane >> 4;
  const int wav  = tid >> 5;
  const int n0   = wav * 16 + col;
  const int8_t* wrow = wb + (size_t)n0 * KKpad;
  const int8_t* arow = s_im + col * KKpad;

  v8i acc = {};
  for (int kb = 0; kb < KKpad; kb += 64) {
    const int8_t* ap = arow + kb + (hi << 4);
    v2i a01 = *(const v2i*)(ap);
    v2i a23 = *(const v2i*)(ap + 8);
    v2i a45 = *(const v2i*)(ap + 32);
    v2i a67 = *(const v2i*)(ap + 40);
    const int8_t* bp = wrow + kb + (hi << 4);
    v4i b03 = *(const v4i*)(bp);
    v4i b47 = *(const v4i*)(bp + 32);

    v8i a = {a01.x, a01.y, a23.x, a23.y, a45.x, a45.y, a67.x, a67.y};
    v8i b = {b03.x, b03.y, b03.z, b03.w, b47.x, b47.y, b47.z, b47.w};
    acc = __builtin_amdgcn_wmma_i32_16x16x64_iu8(true, a, true, b, acc,
                                                 false, false);
  }

  const bool nvalid = n0 < co;
  float bi = nvalid ? bias[n0] : 0.f;
#pragma unroll
  for (int v = 0; v < 8; ++v)
    s_y[wav][v + 8 * hi][col] = (float)acc[v] + bi;
  __syncthreads();

#pragma unroll
  for (int it = 0; it < 3; ++it) {
    int j = hi + 2 * it;
    int p = P0 + j;
    if (nvalid && p < Lp) {
      float m = s_y[wav][2 * j][col];
#pragma unroll
      for (int q = 1; q < 5; ++q) m = fmaxf(m, s_y[wav][2 * j + q][col]);
      pout[((size_t)s * co + n0) * Lp + p] = m;
    }
  }
}

__global__ void mean_kernel(const float* __restrict__ x,
                            float* __restrict__ out, int L, size_t total) {
  size_t idx    = (size_t)blockIdx.x * blockDim.x + threadIdx.x;
  size_t stride = (size_t)gridDim.x * blockDim.x;
  for (; idx < total; idx += stride) {
    const float* base = x + idx * (size_t)L;
    float s = 0.f;
    for (int j = 0; j < L; ++j) s += base[j];
    out[idx] = s / (float)L;
  }
}

extern "C" void kernel_launch(void* const* d_in, const int* in_sizes, int n_in,
                              void* d_out, int out_size, void* d_ws,
                              size_t ws_size, hipStream_t stream) {
  const int CI[7] = {1, 8, 16, 32, 64, 72, 32};
  const int CO[7] = {8, 16, 32, 64, 72, 32, 64};

  const float* x = (const float*)d_in[0];
  auto Wp = [&](int b) { return (const float*)d_in[1 + 4 * b]; };
  auto Bp = [&](int b) { return (const float*)d_in[2 + 4 * b]; };
  auto Gp = [&](int b) { return (const float*)d_in[3 + 4 * b]; };
  auto Ep = [&](int b) { return (const float*)d_in[4 + 4 * b]; };
  auto Mp = [&](int b) { return (const float*)d_in[29 + 2 * b]; };
  auto Vp = [&](int b) { return (const float*)d_in[30 + 2 * b]; };

  char* ws = (char*)d_ws;
  size_t off = 0;
  auto carve = [&](size_t bytes) -> void* {
    off = (off + 255) & ~(size_t)255;
    void* p = ws + off;
    off += bytes;
    return p;
  };
  float*  pbuf0 = (float*)carve(128ull << 20);
  float*  pbuf1 = (float*)carve(128ull << 20);
  int8_t* wball = (int8_t*)carve(1ull << 20);

  const float* cur = x;
  int L = 3600;
  size_t wboff = 0;
  for (int b = 0; b < 7; ++b) {
    const int ci = CI[b], co = CO[b];
    const int co16 = ((co + 15) / 16) * 16;
    const int KK = ci * KW, KKpad = (KK + 63) & ~63;
    int8_t* wbb = wball + wboff;
    wboff += (size_t)co16 * KKpad;

    {
      int total = co16 * KKpad;
      binarize_weights_kernel<<<(total + 255) / 256, 256, 0, stream>>>(
          Wp(b), wbb, co, co16, KK, KKpad);
    }
    const int Lp = (L - 5) / 2 + 1;
    float* pout = (b & 1) ? pbuf1 : pbuf0;
    {
      dim3 grid((Lp + PTILE - 1) / PTILE, BATCH);
      int nwaves = (co + 15) / 16;
      xnor_block_fused_kernel<<<grid, nwaves * 32, 0, stream>>>(
          cur, wbb, Bp(b), Gp(b), Ep(b), Mp(b), Vp(b), pout,
          ci, co, L, Lp, KK, KKpad);
    }
    cur = pout;
    L = Lp;
  }

  {
    size_t total = (size_t)BATCH * 64;
    mean_kernel<<<(int)((total + 255) / 256), 256, 0, stream>>>(
        cur, (float*)d_out, L, total);
  }
}
